// MultiGnns_31653908971900
// MI455X (gfx1250) — compile-verified, matrix operand order checked by reading, not run on hardware
//
#include <hip/hip_runtime.h>
#include <hip/hip_bf16.h>

#define TT 4
#define NA 20000
#define ND 20000
#define EE 100000
#define HH 128
#define LL 2

typedef __attribute__((ext_vector_type(16))) __bf16 v16bf;
typedef __attribute__((ext_vector_type(8)))  __bf16 v8bf;
typedef __attribute__((ext_vector_type(8)))  float  v8f;

__device__ __forceinline__ __bf16 f2bf(float f) {
    unsigned u = __builtin_bit_cast(unsigned, f);
    unsigned r = (u + 0x7FFFu + ((u >> 16) & 1u)) >> 16;   // round-to-nearest-even
    unsigned short s = (unsigned short)r;
    return __builtin_bit_cast(__bf16, s);
}

// Load a 16-element per-lane WMMA fragment (A-row or B-col pattern) from an LDS
// row starting at `base`. ISA 7.12.2 16-bit A layout: VGPR0-3 = K=kb..kb+7,
// VGPR4-7 = K=kb+16..kb+23, kb = (lane>=16) ? 8 : 0. Two aligned 16B loads.
__device__ __forceinline__ v16bf load_frag(const __bf16* base, int ko, int kb) {
    v8bf lo = *(const v8bf*)(base + ko + kb);
    v8bf hi = *(const v8bf*)(base + ko + 16 + kb);
    return __builtin_shufflevector(lo, hi, 0,1,2,3,4,5,6,7,8,9,10,11,12,13,14,15);
}

// -------- Edge message + scatter aggregate:  agg[dst] += relu(x_src[src] + ea) --------
__global__ __launch_bounds__(256)
void edge_agg_kernel(const float* __restrict__ xsrc, const float* __restrict__ ea,
                     const int* __restrict__ src, const int* __restrict__ dst,
                     float* __restrict__ agg, int nE) {
    int gid = blockIdx.x * 256 + threadIdx.x;
    int e = gid >> 5;                 // 32 lanes per edge
    if (e >= nE) return;
    int c = (gid & 31) << 2;          // 4 floats per lane -> 128 cols
    int s = src[e];
    int d = dst[e];
    const float4 xs = *(const float4*)(xsrc + (size_t)s * HH + c);
    const float4 ev = *(const float4*)(ea   + (size_t)e * HH + c);
    float4 m;
    m.x = fmaxf(xs.x + ev.x, 0.f);
    m.y = fmaxf(xs.y + ev.y, 0.f);
    m.z = fmaxf(xs.z + ev.z, 0.f);
    m.w = fmaxf(xs.w + ev.w, 0.f);
    float* p = agg + (size_t)d * HH + c;
    unsafeAtomicAdd(p + 0, m.x);      // global_atomic_add_f32
    unsafeAtomicAdd(p + 1, m.y);
    unsafeAtomicAdd(p + 2, m.z);
    unsafeAtomicAdd(p + 3, m.w);
}

// -------- Fused GINE node update + 2-layer MLP via bf16 WMMA ----------------
// out = [relu]( relu(((1+eps)*xdst + agg) @ W1 + b1) @ W2 + b2 )
// One block = 64 rows. 8 wave32s; wave w owns output columns [16w,16w+16).
__global__ __launch_bounds__(256)
void gine_mlp_kernel(const float* __restrict__ xdst, const float* __restrict__ agg,
                     const float* __restrict__ w1, const float* __restrict__ b1,
                     const float* __restrict__ w2, const float* __restrict__ b2,
                     const float* __restrict__ epsp,
                     float* __restrict__ out, int N, int do_relu) {
    __shared__ __bf16 sX [64][136];    // input tile (bf16), padded rows
    __shared__ __bf16 sW1[128][136];   // W1 transposed: [n][k]
    __shared__ __bf16 sW2[128][136];   // W2 transposed: [n][k]
    __shared__ __bf16 sHm[64][136];    // hidden tile
    __shared__ float  sB1[128];
    __shared__ float  sB2[128];

    const int tid  = threadIdx.x;
    const int row0 = blockIdx.x * 64;
    const float ep = 1.0f + epsp[0];

    // Stage input tile: (1+eps)*x_dst + agg  -> bf16
    for (int i = tid; i < 64 * 128; i += 256) {
        int r = i >> 7, c = i & 127;
        int gr = row0 + r;
        float v = 0.f;
        if (gr < N) v = ep * xdst[(size_t)gr * HH + c] + agg[(size_t)gr * HH + c];
        sX[r][c] = f2bf(v);
    }
    // Stage both weight matrices, transposed (coalesced global reads along n)
    for (int i = tid; i < 128 * 128; i += 256) {
        int k = i >> 7, n = i & 127;
        sW1[n][k] = f2bf(w1[(size_t)k * HH + n]);
        sW2[n][k] = f2bf(w2[(size_t)k * HH + n]);
    }
    if (tid < 128) { sB1[tid] = b1[tid]; sB2[tid] = b2[tid]; }
    __syncthreads();

    const int lane = tid & 31;
    const int wv   = tid >> 5;                    // wave id = N-tile
    const int kb   = (lane & 16) ? 8 : 0;         // K half select
    const int nn   = (wv << 4) + (lane & 15);     // output column (B/C/D)
    const int arow = lane & 15;                   // A row within M-tile
    const int rhi  = (lane & 16) ? 8 : 0;         // C/D row-half select

    // ---- Stage 1: Hmid = relu(X @ W1 + b1) ----
    for (int mt = 0; mt < 4; ++mt) {
        v8f c;
        #pragma unroll
        for (int j = 0; j < 8; ++j) c[j] = sB1[nn];
        #pragma unroll
        for (int ko = 0; ko < 128; ko += 32) {
            v16bf a = load_frag(&sX [mt * 16 + arow][0], ko, kb);
            v16bf b = load_frag(&sW1[nn][0],             ko, kb);
            c = __builtin_amdgcn_wmma_f32_16x16x32_bf16(false, a, false, b,
                                                        (short)0, c, false, false);
        }
        const int rbase = mt * 16 + rhi;
        #pragma unroll
        for (int j = 0; j < 8; ++j) {
            float v = c[j];
            v = v > 0.f ? v : 0.f;
            sHm[rbase + j][nn] = f2bf(v);
        }
    }
    __syncthreads();

    // ---- Stage 2: OUT = Hmid @ W2 + b2  (+optional inter-layer relu) ----
    for (int mt = 0; mt < 4; ++mt) {
        v8f c;
        #pragma unroll
        for (int j = 0; j < 8; ++j) c[j] = sB2[nn];
        #pragma unroll
        for (int ko = 0; ko < 128; ko += 32) {
            v16bf a = load_frag(&sHm[mt * 16 + arow][0], ko, kb);
            v16bf b = load_frag(&sW2[nn][0],             ko, kb);
            c = __builtin_amdgcn_wmma_f32_16x16x32_bf16(false, a, false, b,
                                                        (short)0, c, false, false);
        }
        const int rbase = mt * 16 + rhi;
        #pragma unroll
        for (int j = 0; j < 8; ++j) {
            float v = c[j];
            if (do_relu) v = v > 0.f ? v : 0.f;
            int gr = row0 + rbase + j;
            if (gr < N) out[(size_t)gr * HH + nn] = v;
        }
    }
}

extern "C" void kernel_launch(void* const* d_in, const int* in_sizes, int n_in,
                              void* d_out, int out_size, void* d_ws, size_t ws_size,
                              hipStream_t stream) {
    const float* x_a   = (const float*)d_in[0];
    const float* x_d   = (const float*)d_in[1];
    const float* ea_ad = (const float*)d_in[2];
    const float* ea_da = (const float*)d_in[3];
    const int*   ei_ad = (const int*)  d_in[4];
    const int*   ei_da = (const int*)  d_in[5];
    const float* w1_ad = (const float*)d_in[6];
    const float* b1_ad = (const float*)d_in[7];
    const float* w2_ad = (const float*)d_in[8];
    const float* b2_ad = (const float*)d_in[9];
    const float* ep_ad = (const float*)d_in[10];
    const float* w1_da = (const float*)d_in[11];
    const float* b1_da = (const float*)d_in[12];
    const float* w2_da = (const float*)d_in[13];
    const float* b2_da = (const float*)d_in[14];
    const float* ep_da = (const float*)d_in[15];
    float* out = (float*)d_out;

    float* agg  = (float*)d_ws;                  // max(NA,ND)*H
    float* bufA = agg  + (size_t)NA * HH;        // intermediate x_a
    float* bufD = bufA + (size_t)NA * HH;        // intermediate x_d

    const dim3 eblk(256), egrd((EE * 32 + 255) / 256);
    const dim3 gblk(256);
    const dim3 ggrdD((ND + 63) / 64), ggrdA((NA + 63) / 64);

    for (int t = 0; t < TT; ++t) {
        const float* xa_cur = x_a + (size_t)t * NA * HH;
        const float* xd_cur = x_d + (size_t)t * ND * HH;
        const int* sa = ei_ad + (size_t)t * 2 * EE;       // ad: src row 0, dst row 1
        const int* sd = ei_da + (size_t)t * 2 * EE;
        const float* ea_t = ea_ad + (size_t)t * EE * HH;
        const float* ed_t = ea_da + (size_t)t * EE * HH;

        for (int l = 0; l < LL; ++l) {
            const int last = (l == LL - 1);
            float* xd_nxt = last ? out + (size_t)t * (NA + ND) * HH + (size_t)NA * HH
                                 : bufD;
            float* xa_nxt = last ? out + (size_t)t * (NA + ND) * HH
                                 : bufA;
            const size_t wo = ((size_t)t * LL + l) * HH * HH;
            const size_t bo = ((size_t)t * LL + l) * HH;
            const size_t eo = (size_t)t * LL + l;

            // a->d conv: xd_new = MLP_ad((1+eps)*xd + sum relu(xa[src]+ea_ad))
            hipMemsetAsync(agg, 0, (size_t)ND * HH * sizeof(float), stream);
            edge_agg_kernel<<<egrd, eblk, 0, stream>>>(xa_cur, ea_t, sa, sa + EE, agg, EE);
            gine_mlp_kernel<<<ggrdD, gblk, 0, stream>>>(xd_cur, agg,
                w1_ad + wo, b1_ad + bo, w2_ad + wo, b2_ad + bo, ep_ad + eo,
                xd_nxt, ND, last ? 0 : 1);

            // d->a conv: xa_new = MLP_da((1+eps)*xa + sum relu(xd[src]+ea_da))
            hipMemsetAsync(agg, 0, (size_t)NA * HH * sizeof(float), stream);
            edge_agg_kernel<<<egrd, eblk, 0, stream>>>(xd_cur, ed_t, sd, sd + EE, agg, EE);
            gine_mlp_kernel<<<ggrdA, gblk, 0, stream>>>(xa_cur, agg,
                w1_da + wo, b1_da + bo, w2_da + wo, b2_da + bo, ep_da + eo,
                xa_nxt, NA, last ? 0 : 1);

            xa_cur = xa_nxt;
            xd_cur = xd_nxt;
        }
    }
}
